// BaselineEncoder_67190468379233
// MI455X (gfx1250) — matrix operand order repaired and checked by reading, not compiled, not run on hardware
//
#include <hip/hip_runtime.h>
#include <hip/hip_bf16.h>

typedef __bf16 bf16_t;
typedef __bf16 v16bf __attribute__((ext_vector_type(16)));
typedef float  v8f   __attribute__((ext_vector_type(8)));
typedef unsigned v4u __attribute__((ext_vector_type(4)));
typedef int    v4i   __attribute__((ext_vector_type(4)));
typedef int    v8i   __attribute__((ext_vector_type(8)));

#define BN_EPS 1e-5f

__device__ __forceinline__ bf16_t f2bf(float f) {
    union { float f; unsigned u; } a; a.f = f;
    unsigned u = a.u;
    unsigned r = (u + 0x7FFFu + ((u >> 16) & 1u)) >> 16;
    union { unsigned short s; bf16_t b; } c; c.s = (unsigned short)r;
    return c.b;
}

union Frag16 { v16bf v; v4u q[2]; unsigned u[8]; };

__device__ __forceinline__ v8f wmma_bf16(v16bf a, v16bf b, v8f c) {
    return __builtin_amdgcn_wmma_f32_16x16x32_bf16(false, a, false, b, (short)0, c, false, false);
}

// ---------------------------------------------------------------------------
// f32 -> bf16 conversion
// ---------------------------------------------------------------------------
__global__ void cvt_f32_bf16(const float* __restrict__ in, bf16_t* __restrict__ out, int n) {
    int i = blockIdx.x * blockDim.x + threadIdx.x;
    if (i < n) out[i] = f2bf(in[i]);
}

// ---------------------------------------------------------------------------
// BF16 WMMA GEMM:  C[M,N] = A[M,K] (row major) * B[K,N] (row major)
// Block tile 128x128x32, 256 threads = 8 waves in a 4(M) x 2(N) grid,
// each wave owns 2x4 = 8 wmma 16x16 tiles.
// A tile is staged into LDS by the Tensor Data Mover (tensor_load_to_lds,
// one D# per K-step, wave 0 issues, TENSORcnt-synced); the B tile needs a
// 16-bit transpose so it is staged manually.
// EPI: 0 = store bf16, 1 = relu -> bf16, 2 = (+resid, BN) -> f32 + bf16,
//      3 = (+resid, BN) -> f32 only
// ---------------------------------------------------------------------------
template<int EPI>
__global__ __launch_bounds__(256)
void gemm_bf16_wmma(const bf16_t* __restrict__ A,
                    const bf16_t* __restrict__ Bm, long strideB,
                    float* __restrict__ outF, long strideOF,
                    bf16_t* __restrict__ outB, long strideOB,
                    const float* __restrict__ resid, long strideR,
                    const float* __restrict__ gamma, const float* __restrict__ beta,
                    const float* __restrict__ mean,  const float* __restrict__ var,
                    int M, int N, int K)
{
    constexpr int BM = 128, BNt = 128, BK = 32;
    __shared__ __align__(16) bf16_t As[BM * BK];        // [m][k]  (TDM-filled)
    __shared__ __align__(16) bf16_t Bs[BNt * BK];       // transposed: [n][k]

    const int t    = threadIdx.x;
    const int wave = t >> 5;
    const int lane = t & 31;
    const int hf   = lane >> 4;
    const int ln   = lane & 15;
    const int wm   = wave & 3;   // M-dir (32 rows each)
    const int wn   = wave >> 2;  // N-dir (64 cols each)
    const int m0   = blockIdx.y * BM;
    const int n0   = blockIdx.x * BNt;
    const int bz   = blockIdx.z;

    const bf16_t* Bp = Bm + (long)bz * strideB;

    // ---- constant parts of the TDM descriptor (wave-uniform) ----
    const unsigned lds_as = (unsigned)(unsigned long long)(void*)&As[0];
    v8i g1;
    g1[0] = 0x00010000;                                   // data_size=1 (2B), mask=0
    g1[1] = (int)(((unsigned)K & 0xFFFFu) << 16);         // tensor_dim0 lo16
    g1[2] = (int)((((unsigned)M & 0xFFFFu) << 16) | ((unsigned)K >> 16)); // dim1 lo | dim0 hi
    g1[3] = (int)((32u << 16) | ((unsigned)M >> 16));     // tile_dim0=32 | dim1 hi
    g1[4] = (int)128;                                     // tile_dim1=128, tile_dim2=0
    g1[5] = (int)(unsigned)K;                             // tensor_dim0_stride lo32
    g1[6] = 0;                                            // stride0 hi16 | stride1 lo16
    g1[7] = 0;
    v4i gz4; gz4[0] = 0; gz4[1] = 0; gz4[2] = 0; gz4[3] = 0; // groups 2/3 (2-D tensor)
    v8i gz8;
#pragma unroll
    for (int i = 0; i < 8; i++) gz8[i] = 0;

    v8f acc[2][4];
#pragma unroll
    for (int i = 0; i < 2; i++)
#pragma unroll
        for (int j = 0; j < 4; j++)
#pragma unroll
            for (int v = 0; v < 8; v++) acc[i][j][v] = 0.0f;

    for (int k0 = 0; k0 < K; k0 += BK) {
        if (k0 + BK < K)
            __builtin_prefetch((const void*)(Bp + (long)(k0 + BK) * N + n0), 0, 1);
        __syncthreads();   // everyone finished reading As/Bs of previous step

        // ---- A tile 128x32 via Tensor Data Mover (wave 0 issues) ----
        if (wave == 0) {
            unsigned long long ga =
                (unsigned long long)(const void*)(A + (long)m0 * K + k0);
            v4u g0;
            g0[0] = 1u;                                    // count=1
            g0[1] = lds_as;                                // lds_addr
            g0[2] = (unsigned)(ga & 0xFFFFFFFFu);          // global_addr[31:0]
            g0[3] = (unsigned)((ga >> 32) & 0x01FFFFFFu) | 0x80000000u; // addr[56:32] | type=2
            __builtin_amdgcn_tensor_load_to_lds(g0, g1, gz4, gz4, gz8, 0);
        }

        // ---- B tile 32x128, transposed into Bs[n][k] (all waves) ----
#pragma unroll
        for (int i = 0; i < 2; i++) {
            int idx = (t + i * 256) * 8;
            int row = idx >> 7, col = idx & 127;
            union { v4u q; unsigned short s[8]; } d;
            d.q = *(const v4u*)(Bp + (long)(k0 + row) * N + n0 + col);
#pragma unroll
            for (int j = 0; j < 8; j++)
                Bs[(col + j) * BK + row] = ((const bf16_t*)d.s)[j];
        }

        if (wave == 0) __builtin_amdgcn_s_wait_tensorcnt(0);
        __syncthreads();   // TDM data + Bs visible to all waves

        // A fragments: lane row = ln, elem pair p -> k per ISA 16-bit A layout
        Frag16 af[2];
#pragma unroll
        for (int i = 0; i < 2; i++) {
            const bf16_t* base = &As[(wm * 32 + i * 16 + ln) * BK];
#pragma unroll
            for (int p = 0; p < 8; p++) {
                int k = (p < 4) ? (2 * p + 8 * hf) : (16 + 2 * (p - 4) + 8 * hf);
                af[i].u[p] = *(const unsigned*)(base + k);
            }
        }
        // B fragments: lane col = ln, elems k = 8*hf + e (e<8), 16 + 8*hf + (e-8) (e>=8) in Bs[n][k]
        Frag16 bfr[4];
#pragma unroll
        for (int j = 0; j < 4; j++) {
            const bf16_t* base = &Bs[(wn * 64 + j * 16 + ln) * BK + 8 * hf];
            bfr[j].q[0] = *(const v4u*)(base);
            bfr[j].q[1] = *(const v4u*)(base + 16);
        }
#pragma unroll
        for (int i = 0; i < 2; i++)
#pragma unroll
            for (int j = 0; j < 4; j++)
                acc[i][j] = wmma_bf16(af[i].v, bfr[j].v, acc[i][j]);
    }

    float*        oF = outF  ? outF  + (long)bz * strideOF : (float*)0;
    bf16_t*       oB = outB  ? outB  + (long)bz * strideOB : (bf16_t*)0;
    const float*  rs = resid ? resid + (long)bz * strideR  : (const float*)0;

#pragma unroll
    for (int i = 0; i < 2; i++) {
#pragma unroll
        for (int j = 0; j < 4; j++) {
#pragma unroll
            for (int v = 0; v < 8; v++) {
                int  mg  = m0 + wm * 32 + i * 16 + v + 8 * hf;   // C/D row = vgpr + 8*half
                int  ng  = n0 + wn * 64 + j * 16 + ln;           // C/D col = lane&15
                long off = (long)mg * N + ng;
                float val = acc[i][j][v];
                if (EPI == 0) {
                    oB[off] = f2bf(val);
                } else if (EPI == 1) {
                    oB[off] = f2bf(fmaxf(val, 0.0f));
                } else {
                    float inv = gamma[mg] * rsqrtf(var[mg] + BN_EPS);
                    float sh  = beta[mg] - mean[mg] * inv;
                    float y   = (val + rs[off]) * inv + sh;
                    oF[off] = y;
                    if (EPI == 2) oB[off] = f2bf(y);
                }
            }
        }
    }
}

// ---------------------------------------------------------------------------
// Flash attention, one (batch, head) per blockIdx.{z,y}, 128 queries per block.
// qkv layout (bf16): [B][3*C][N]; q rows = h*64+d, k rows = 512+h*64+d,
// v rows = 1024+h*64+d.  Each wave owns 16 query rows; keys streamed in
// chunks of 64 with online softmax (wave32 shfl_xor within 16-lane halves).
// ---------------------------------------------------------------------------
__global__ __launch_bounds__(256)
void attn_flash(const bf16_t* __restrict__ qkv, bf16_t* __restrict__ out, int Nseq)
{
    constexpr int D = 64;
    __shared__ __align__(16) bf16_t QT[128 * D];       // [n][d]
    __shared__ __align__(16) bf16_t KT[64 * D];        // [m][d]
    __shared__ __align__(16) bf16_t Pl[8 * 16 * 64];   // per-wave P [row][m]

    const int t    = threadIdx.x;
    const int wave = t >> 5;
    const int lane = t & 31;
    const int hf   = lane >> 4;
    const int ln   = lane & 15;
    const int n0   = blockIdx.x * 128;
    const int h    = blockIdx.y;
    const int b    = blockIdx.z;
    const long Ns  = (long)Nseq;

    const bf16_t* qp = qkv + ((long)b * 1536 +        h * 64) * Ns;
    const bf16_t* kp = qkv + ((long)b * 1536 +  512 + h * 64) * Ns;
    const bf16_t* vp = qkv + ((long)b * 1536 + 1024 + h * 64) * Ns;

    // stage QT[128][64] (transpose from (d,n) storage)
#pragma unroll
    for (int i = 0; i < 4; i++) {
        int idx = (t + i * 256) * 8;
        int d = idx >> 7, n = idx & 127;
        union { v4u q; unsigned short s[8]; } u;
        u.q = *(const v4u*)(qp + (long)d * Ns + n0 + n);
#pragma unroll
        for (int j = 0; j < 8; j++) QT[(n + j) * D + d] = ((const bf16_t*)u.s)[j];
    }

    float m_run[8], l_run[8];
    v8f o_acc[4];
#pragma unroll
    for (int v = 0; v < 8; v++) { m_run[v] = -1e30f; l_run[v] = 0.0f; }
#pragma unroll
    for (int j = 0; j < 4; j++)
#pragma unroll
        for (int v = 0; v < 8; v++) o_acc[j][v] = 0.0f;

    const float scale = 0.125f;   // 1/sqrt(64)

    for (int mc = 0; mc < Nseq; mc += 64) {
        __syncthreads();
        // stage KT[64][64] transposed
#pragma unroll
        for (int i = 0; i < 2; i++) {
            int idx = (t + i * 256) * 8;
            int d = idx >> 6, m = idx & 63;
            union { v4u q; unsigned short s[8]; } u;
            u.q = *(const v4u*)(kp + (long)d * Ns + mc + m);
#pragma unroll
            for (int j = 0; j < 8; j++) KT[(m + j) * D + d] = ((const bf16_t*)u.s)[j];
        }
        __syncthreads();

        // S = Q^T K : 16 rows x 64 keys per wave
        v8f s[4];
#pragma unroll
        for (int j = 0; j < 4; j++)
#pragma unroll
            for (int v = 0; v < 8; v++) s[j][v] = 0.0f;

#pragma unroll
        for (int kk = 0; kk < D; kk += 32) {
            Frag16 aq;
            const bf16_t* abase = &QT[(wave * 16 + ln) * D + kk];
#pragma unroll
            for (int p = 0; p < 8; p++) {
                int k = (p < 4) ? (2 * p + 8 * hf) : (16 + 2 * (p - 4) + 8 * hf);
                aq.u[p] = *(const unsigned*)(abase + k);
            }
#pragma unroll
            for (int j = 0; j < 4; j++) {
                Frag16 bk;
                const bf16_t* bbase = &KT[(j * 16 + ln) * D + kk + 8 * hf];
                bk.q[0] = *(const v4u*)(bbase);
                bk.q[1] = *(const v4u*)(bbase + 16);
                s[j] = wmma_bf16(aq.v, bk.v, s[j]);
            }
        }

        // online softmax, row r = v + 8*hf, reduce over 16-lane half groups
#pragma unroll
        for (int v = 0; v < 8; v++) {
            float rmax = -1e30f;
#pragma unroll
            for (int j = 0; j < 4; j++) rmax = fmaxf(rmax, s[j][v] * scale);
#pragma unroll
            for (int x = 1; x < 16; x <<= 1) rmax = fmaxf(rmax, __shfl_xor(rmax, x, 32));
            float mnew  = fmaxf(m_run[v], rmax);
            float alpha = __expf(m_run[v] - mnew);
            float rsum  = 0.0f;
#pragma unroll
            for (int j = 0; j < 4; j++) {
                float p = __expf(s[j][v] * scale - mnew);
                s[j][v] = p;
                rsum += p;
            }
#pragma unroll
            for (int x = 1; x < 16; x <<= 1) rsum += __shfl_xor(rsum, x, 32);
            l_run[v] = l_run[v] * alpha + rsum;
            m_run[v] = mnew;
#pragma unroll
            for (int j = 0; j < 4; j++) o_acc[j][v] *= alpha;
        }

        // write P (D-layout -> LDS row major) then reload as A fragments
        bf16_t* Pw = &Pl[wave * 16 * 64];
#pragma unroll
        for (int j = 0; j < 4; j++)
#pragma unroll
            for (int v = 0; v < 8; v++)
                Pw[(v + 8 * hf) * 64 + j * 16 + ln] = f2bf(s[j][v]);
        __syncthreads();

        // O += P * V^T   (V read straight from global: K-contiguous along m)
#pragma unroll
        for (int kk = 0; kk < 64; kk += 32) {
            Frag16 ap;
            const bf16_t* abase = &Pw[ln * 64 + kk];
#pragma unroll
            for (int p = 0; p < 8; p++) {
                int k = (p < 4) ? (2 * p + 8 * hf) : (16 + 2 * (p - 4) + 8 * hf);
                ap.u[p] = *(const unsigned*)(abase + k);
            }
#pragma unroll
            for (int j = 0; j < 4; j++) {
                Frag16 bv;
                const bf16_t* bbase = vp + (long)(j * 16 + ln) * Ns + mc + kk + 8 * hf;
                bv.q[0] = *(const v4u*)(bbase);
                bv.q[1] = *(const v4u*)(bbase + 16);
                o_acc[j] = wmma_bf16(ap.v, bv.v, o_acc[j]);
            }
        }
    }

    // finalize: out[b, h*64 + d, n] = O / l
    bf16_t* ob = out + ((long)b * 512 + h * 64) * Ns;
#pragma unroll
    for (int j = 0; j < 4; j++) {
#pragma unroll
        for (int v = 0; v < 8; v++) {
            float val = o_acc[j][v] / l_run[v];
            int d = j * 16 + ln;
            int n = n0 + wave * 16 + v + 8 * hf;
            ob[(long)d * Ns + n] = f2bf(val);
        }
    }
}

// ---------------------------------------------------------------------------
// host-side orchestration
// ---------------------------------------------------------------------------
extern "C" void kernel_launch(void* const* d_in, const int* in_sizes, int n_in,
                              void* d_out, int out_size, void* d_ws, size_t ws_size,
                              hipStream_t stream) {
    const float* x      = (const float*)d_in[0];
    const float* w_qkv  = (const float*)d_in[1];
    const float* w_proj = (const float*)d_in[2];
    const float* w_fc1  = (const float*)d_in[3];
    const float* w_fc2  = (const float*)d_in[4];
    const float* gamma  = (const float*)d_in[5];
    const float* beta   = (const float*)d_in[6];
    const float* mean   = (const float*)d_in[7];
    const float* var    = (const float*)d_in[8];
    float* out = (float*)d_out;
    (void)in_sizes; (void)n_in; (void)out_size; (void)ws_size;

    const int Bb = 4, C = 512, Nn = 2048, Hh = 1024;

    char*  ws  = (char*)d_ws;
    size_t off = 0;
    auto take = [&](size_t bytes) -> char* {
        char* p = ws + off;
        off = (off + bytes + 255) & ~(size_t)255;
        return p;
    };

    bf16_t* x_bf    = (bf16_t*)take((size_t)Bb * C * Nn * 2);
    bf16_t* wqkv_bf = (bf16_t*)take((size_t)3 * C * C * 2);
    bf16_t* wprj_bf = (bf16_t*)take((size_t)C * C * 2);
    bf16_t* wf1_bf  = (bf16_t*)take((size_t)Hh * C * 2);
    bf16_t* wf2_bf  = (bf16_t*)take((size_t)C * Hh * 2);
    bf16_t* qkv_bf  = (bf16_t*)take((size_t)Bb * 3 * C * Nn * 2);
    bf16_t* ao_bf   = (bf16_t*)take((size_t)Bb * C * Nn * 2);
    float*  y1_f    = (float*) take((size_t)Bb * C * Nn * 4);
    bf16_t* y1_bf   = (bf16_t*)take((size_t)Bb * C * Nn * 2);
    bf16_t* h_bf    = (bf16_t*)take((size_t)Bb * Hh * Nn * 2);

    // f32 -> bf16 conversions
    int n;
    n = Bb * C * Nn; cvt_f32_bf16<<<(n + 255) / 256, 256, 0, stream>>>(x,      x_bf,    n);
    n = 3 * C * C;   cvt_f32_bf16<<<(n + 255) / 256, 256, 0, stream>>>(w_qkv,  wqkv_bf, n);
    n = C * C;       cvt_f32_bf16<<<(n + 255) / 256, 256, 0, stream>>>(w_proj, wprj_bf, n);
    n = Hh * C;      cvt_f32_bf16<<<(n + 255) / 256, 256, 0, stream>>>(w_fc1,  wf1_bf,  n);
    n = C * Hh;      cvt_f32_bf16<<<(n + 255) / 256, 256, 0, stream>>>(w_fc2,  wf2_bf,  n);

    dim3 blk(256);

    // qkv = w_qkv @ x          (M=1536, K=512, N=2048, batched over B)
    gemm_bf16_wmma<0><<<dim3(Nn / 128, (3 * C) / 128, Bb), blk, 0, stream>>>(
        wqkv_bf, x_bf, (long)C * Nn,
        (float*)0, 0, qkv_bf, (long)3 * C * Nn,
        (const float*)0, 0, (const float*)0, (const float*)0, (const float*)0, (const float*)0,
        3 * C, Nn, C);

    // flash attention
    attn_flash<<<dim3(Nn / 128, 8, Bb), blk, 0, stream>>>(qkv_bf, ao_bf, Nn);

    // y1 = BN(x + w_proj @ attn_out)   -> f32 + bf16
    gemm_bf16_wmma<2><<<dim3(Nn / 128, C / 128, Bb), blk, 0, stream>>>(
        wprj_bf, ao_bf, (long)C * Nn,
        y1_f, (long)C * Nn, y1_bf, (long)C * Nn,
        x, (long)C * Nn, gamma, beta, mean, var,
        C, Nn, C);

    // h = relu(w_fc1 @ y1)     -> bf16
    gemm_bf16_wmma<1><<<dim3(Nn / 128, Hh / 128, Bb), blk, 0, stream>>>(
        wf1_bf, y1_bf, (long)C * Nn,
        (float*)0, 0, h_bf, (long)Hh * Nn,
        (const float*)0, 0, (const float*)0, (const float*)0, (const float*)0, (const float*)0,
        Hh, Nn, C);

    // out = BN(y1 + w_fc2 @ h) -> f32 (d_out)
    gemm_bf16_wmma<3><<<dim3(Nn / 128, C / 128, Bb), blk, 0, stream>>>(
        wf2_bf, h_bf, (long)Hh * Nn,
        out, (long)C * Nn, (bf16_t*)0, 0,
        y1_f, (long)C * Nn, gamma, beta, mean, var,
        C, Nn, Hh);
}
